// RegionProposalNetwork_66288525247249
// MI455X (gfx1250) — matrix operand order repaired and checked by reading, not compiled, not run on hardware
//
#include <hip/hip_runtime.h>
#include <hip/hip_bf16.h>
#include <math.h>

// ---------------------------------------------------------------------------
// RPN forward for MI455X (gfx1250, wave32, WMMA).
// Dominant cost: 3x3 conv 1024->512 over 75x120 = 85 GFLOP -> f16 WMMA GEMM.
// ---------------------------------------------------------------------------

typedef __attribute__((ext_vector_type(16))) _Float16 v16h;
typedef __attribute__((ext_vector_type(8)))  _Float16 v8h;
typedef __attribute__((ext_vector_type(8)))  float    v8f;

#define FEAT_H 75
#define FEAT_W 120
#define NPIX   9000            // 75*120
#define C_IN   1024
#define C_MID  512
#define NQ     288             // K chunks: 9216 / 32
#define HALO_H 77
#define HALO_W 122
#define NANCH  9
#define NROWS  81000           // NPIX * NANCH
#define PRE_N  6000
#define POST_N 1000
#define SORT_N 131072          // next pow2 >= 81000

// d_out float offsets (total 814000 floats)
#define A_OFF    0
#define OBJ_OFF  324000
#define TR_OFF   486000
#define PROP_OFF 810000

// ---------------------------------------------------------------------------
// 1) Pack features fp32 NCHW -> f16 HWC with 1-px zero halo: F[77][122][1024]
// ---------------------------------------------------------------------------
__global__ void pack_features(const float* __restrict__ feat, _Float16* __restrict__ F) {
    int id = blockIdx.x * blockDim.x + threadIdx.x;
    const int total = HALO_H * HALO_W * C_IN;
    if (id >= total) return;
    int c  = id & (C_IN - 1);
    int sp = id >> 10;
    int x  = sp % HALO_W;
    int y  = sp / HALO_W;
    float v = 0.0f;
    if (y >= 1 && y <= FEAT_H && x >= 1 && x <= FEAT_W)
        v = feat[c * NPIX + (y - 1) * FEAT_W + (x - 1)];
    F[id] = (_Float16)v;
}

// ---------------------------------------------------------------------------
// 2) Pack w_feat (512,1024,3,3) fp32 into the exact B-fragment VGPR layout:
//    Wp[((ct*NQ + q)*32 + lane)*16 + e]   ct = out-channel tile (16 ch),
//    q = K chunk of 32, lane half selects K {0..7,16..23} / {8..15,24..31},
//    K ordering: k = (kh*3+kw)*1024 + c.
// ---------------------------------------------------------------------------
__global__ void pack_weights(const float* __restrict__ w, _Float16* __restrict__ Wp) {
    int id = blockIdx.x * blockDim.x + threadIdx.x;
    const int total = 32 * NQ * 32 * 16;   // 512 ch-tiles*16 x 9216 = 4.7M
    if (id >= total) return;
    int e    = id & 15;
    int rest = id >> 4;
    int lane = rest & 31; rest >>= 5;
    int q    = rest % NQ;
    int ct   = rest / NQ;                    // 0..31
    int Klocal = ((e >> 3) << 4) + ((lane >> 4) << 3) + (e & 7);     // 16*(e>=8)+8*(lane>=16)+(e&7)
    int k   = q * 32 + Klocal;
    int seg = k >> 10;                       // 0..8
    int c   = k & 1023;
    int kh  = seg / 3, kw = seg - kh * 3;
    int oc  = ct * 16 + (lane & 15);
    Wp[id] = (_Float16)w[(oc * C_IN + c) * 9 + kh * 3 + kw];
}

// ---------------------------------------------------------------------------
// 3) Anchors (analytic) -> d_out[A_OFF ..)
// ---------------------------------------------------------------------------
__global__ void anchors_kernel(float* __restrict__ out,
                               const int* __restrict__ iw, const int* __restrict__ ih) {
    int id = blockIdx.x * blockDim.x + threadIdx.x;
    if (id >= NROWS) return;
    float W = (float)iw[0], H = (float)ih[0];
    int a  = id % NANCH;
    int px = id / NANCH;
    int ri = a / 3, si = a % 3;
    int iy = px / FEAT_W, ix = px % FEAT_W;
    float cy = H * (float)(iy + 1) / 76.0f;   // linspace(0,H,77)[1:-1]
    float cx = W * (float)(ix + 1) / 121.0f;  // linspace(0,W,122)[1:-1]
    const float rs[3] = {0.5f, 1.0f, 2.0f};
    const float ss[3] = {128.0f, 256.0f, 512.0f};
    float r = rs[ri], s = ss[si];
    float w = s * sqrtf(1.0f / r);
    float h = s * sqrtf(r);
    out[A_OFF + id * 4 + 0] = cx - 0.5f * w;
    out[A_OFF + id * 4 + 1] = cy - 0.5f * h;
    out[A_OFF + id * 4 + 2] = cx + 0.5f * w;
    out[A_OFF + id * 4 + 3] = cy + 0.5f * h;
}

// ---------------------------------------------------------------------------
// 4) 3x3 conv as implicit GEMM with v_wmma_f32_16x16x32_f16.
//    Wave tile: 64 pixels x 64 out-channels (4x4 frags, 16 accumulators).
//    Block = 8 waves sharing one 64-pixel tile (A reuse), covering 512 ch.
//    Grid = 141 blocks (141*64 = 9024 >= 9000 pixels).
// ---------------------------------------------------------------------------
__global__ __launch_bounds__(256)
void conv3x3_wmma(const _Float16* __restrict__ F, const _Float16* __restrict__ Wp,
                  const float* __restrict__ bfeat, float* __restrict__ mid) {
    const int lane  = threadIdx.x & 31;
    const int wv    = threadIdx.x >> 5;
    const int gw    = blockIdx.x * 8 + wv;
    const int pxT   = gw >> 3;       // 0..140 (same for all 8 waves of block)
    const int chT   = gw & 7;        // 0..7 -> 64 channels each
    const int pxbase = pxT * 64;
    const int lhalf  = lane >> 4;    // 0/1
    const int lmod   = lane & 15;
    const int ctbase = chT * 4;      // 16-channel tile index base

    // pixel coords for the 4 A fragments (clamped for safe addressing)
    int hh[4], ww[4];
#pragma unroll
    for (int i = 0; i < 4; ++i) {
        int p = pxbase + i * 16 + lmod;
        if (p > NPIX - 1) p = NPIX - 1;
        hh[i] = p / FEAT_W;
        ww[i] = p % FEAT_W;
    }

    v8f acc[4][4];
#pragma unroll
    for (int i = 0; i < 4; ++i)
#pragma unroll
        for (int j = 0; j < 4; ++j) acc[i][j] = v8f{};

    for (int q = 0; q < NQ; ++q) {
        const int seg = q >> 5;               // 0..8 -> (kh,kw)
        const int c0  = (q & 31) << 5;        // channel base of this chunk
        const int kh  = seg / 3, kw = seg - kh * 3;
        const int cA  = c0 + lhalf * 8;

        // A fragments: 16x32 f16, lane<16 holds K{0..7,16..23}, lane>=16 +8.
        v16h a[4];
#pragma unroll
        for (int i = 0; i < 4; ++i) {
            const _Float16* base =
                F + (((hh[i] + kh) * HALO_W + (ww[i] + kw)) * C_IN + cA);
            v8h lo = *(const v8h*)(base);        // K e=0..7
            v8h hi = *(const v8h*)(base + 16);   // K e=8..15 (+16 channels)
            v16h t;
#pragma unroll
            for (int e = 0; e < 8; ++e) { t[e] = lo[e]; t[e + 8] = hi[e]; }
            a[i] = t;
        }

        // B fragments: pre-swizzled contiguous 32B per lane.
        v16h b[4];
#pragma unroll
        for (int j = 0; j < 4; ++j)
            b[j] = *(const v16h*)(Wp + (((ctbase + j) * NQ + q) * 32 + lane) * 16);

        if (q + 1 < NQ)   // gfx1250 global_prefetch_b8 for next weight chunk
            __builtin_prefetch(Wp + ((ctbase * NQ + q + 1) * 32 + lane) * 16, 0, 0);

#pragma unroll
        for (int i = 0; i < 4; ++i)
#pragma unroll
            for (int j = 0; j < 4; ++j)
                acc[i][j] = __builtin_amdgcn_wmma_f32_16x16x32_f16(
                    false, a[i], false, b[j], (short)0, acc[i][j], false, false);
    }

    // Epilogue: bias + ReLU, coalesced store to mid[px][512].
#pragma unroll
    for (int j = 0; j < 4; ++j) {
        const int ch = (ctbase + j) * 16 + lmod;
        const float bb = bfeat[ch];
#pragma unroll
        for (int i = 0; i < 4; ++i) {
#pragma unroll
            for (int r = 0; r < 8; ++r) {
                int px = pxbase + i * 16 + r + lhalf * 8;   // D: M = r + 8*(lane>=16)
                if (px < NPIX) {
                    float v = acc[i][j][r] + bb;
                    mid[px * C_MID + ch] = v > 0.0f ? v : 0.0f;
                }
            }
        }
    }
}

// ---------------------------------------------------------------------------
// 5) 1x1 convs (obj 18ch, trans 36ch) -> d_out, LDS-staged per pixel.
// ---------------------------------------------------------------------------
__global__ __launch_bounds__(64)
void conv1x1(const float* __restrict__ mid,
             const float* __restrict__ wobj, const float* __restrict__ bobj,
             const float* __restrict__ wtr,  const float* __restrict__ btr,
             float* __restrict__ out) {
    __shared__ float s[C_MID];
    const int px = blockIdx.x;
    const float* m = mid + px * C_MID;
    for (int c = threadIdx.x; c < C_MID; c += 64) s[c] = m[c];
    __syncthreads();
    const int t = threadIdx.x;
    if (t < 18) {
        const float* wp = wobj + t * C_MID;
        float acc = bobj[t];
        for (int c = 0; c < C_MID; ++c) acc += wp[c] * s[c];
        out[OBJ_OFF + px * 18 + t] = acc;
    } else if (t < 54) {
        const int o = t - 18;
        const float* wp = wtr + o * C_MID;
        float acc = btr[o];
        for (int c = 0; c < C_MID; ++c) acc += wp[c] * s[c];
        out[TR_OFF + px * 36 + o] = acc;
    }
}

// ---------------------------------------------------------------------------
// 6) Top-k via full bitonic sort (descending score, tie -> lower index).
// ---------------------------------------------------------------------------
__global__ void sort_init(const float* __restrict__ out,
                          float* __restrict__ key, int* __restrict__ val) {
    int i = blockIdx.x * blockDim.x + threadIdx.x;
    if (i >= SORT_N) return;
    if (i < NROWS) { key[i] = out[OBJ_OFF + i * 2 + 1]; val[i] = i; }
    else           { key[i] = -3.4e38f;                 val[i] = i; }
}

__global__ void bitonic_step(float* __restrict__ key, int* __restrict__ val,
                             int j, int k) {
    int i = blockIdx.x * blockDim.x + threadIdx.x;
    int ixj = i ^ j;
    if (ixj <= i || i >= SORT_N) return;
    float ka = key[i], kb = key[ixj];
    int   va = val[i], vb = val[ixj];
    // b_ij: "i before ixj" is already the desired (descending) order
    bool b_ij   = (ka > kb) || (ka == kb && va < vb);
    bool doSwap = ((i & k) == 0) ? !b_ij : b_ij;
    if (doSwap) { key[i] = kb; key[ixj] = ka; val[i] = vb; val[ixj] = va; }
}

// ---------------------------------------------------------------------------
// 7) Gather top 6000, apply box transform, clip; SoA + areas for NMS.
// ---------------------------------------------------------------------------
__global__ void gather_transform(const int* __restrict__ val,
                                 const float* __restrict__ out,
                                 float* __restrict__ bx, int* __restrict__ keep,
                                 const int* __restrict__ iw, const int* __restrict__ ih) {
    int t = blockIdx.x * blockDim.x + threadIdx.x;
    if (t >= PRE_N) return;
    int idx = val[t];
    const float* A = out + A_OFF + idx * 4;
    const float* T = out + TR_OFF + idx * 4;
    float x1 = A[0], y1 = A[1], x2 = A[2], y2 = A[3];
    float w = x2 - x1, h = y2 - y1;
    float cx = x1 + 0.5f * w + T[0] * w;
    float cy = y1 + 0.5f * h + T[1] * h;
    float w2 = expf(T[2]) * w;
    float h2 = expf(T[3]) * h;
    float W = (float)iw[0], H = (float)ih[0];
    float nx1 = fminf(fmaxf(cx - 0.5f * w2, 0.0f), W);
    float ny1 = fminf(fmaxf(cy - 0.5f * h2, 0.0f), H);
    float nx2 = fminf(fmaxf(cx + 0.5f * w2, 0.0f), W);
    float ny2 = fminf(fmaxf(cy + 0.5f * h2, 0.0f), H);
    bx[t]             = nx1;
    bx[PRE_N + t]     = ny1;
    bx[2 * PRE_N + t] = nx2;
    bx[3 * PRE_N + t] = ny2;
    bx[4 * PRE_N + t] = (nx2 - nx1 + 1.0f) * (ny2 - ny1 + 1.0f);
    keep[t] = 1;
}

// ---------------------------------------------------------------------------
// 8) Sequential-semantics NMS, single block, keep flags in LDS.
// ---------------------------------------------------------------------------
__global__ __launch_bounds__(1024)
void nms_kernel(const float* __restrict__ bx, int* __restrict__ keep) {
    __shared__ unsigned char ks[PRE_N];
    for (int i = threadIdx.x; i < PRE_N; i += 1024) ks[i] = 1;
    const float* X1 = bx;
    const float* Y1 = bx + PRE_N;
    const float* X2 = bx + 2 * PRE_N;
    const float* Y2 = bx + 3 * PRE_N;
    const float* AR = bx + 4 * PRE_N;
    for (int i = 0; i < PRE_N; ++i) {
        __syncthreads();
        if (ks[i]) {
            float x1 = X1[i], y1 = Y1[i], x2 = X2[i], y2 = Y2[i], ai = AR[i];
            for (int j = i + 1 + (int)threadIdx.x; j < PRE_N; j += 1024) {
                float xx1 = fmaxf(x1, X1[j]), yy1 = fmaxf(y1, Y1[j]);
                float xx2 = fminf(x2, X2[j]), yy2 = fminf(y2, Y2[j]);
                float inter = fmaxf(xx2 - xx1 + 1.0f, 0.0f) *
                              fmaxf(yy2 - yy1 + 1.0f, 0.0f);
                float iou = inter / (ai + AR[j] - inter);
                if (iou > 0.7f) ks[j] = 0;
            }
        }
    }
    __syncthreads();
    for (int i = threadIdx.x; i < PRE_N; i += 1024) keep[i] = ks[i];
}

// ---------------------------------------------------------------------------
// 9) Compact kept boxes (score order == index order), zero-pad to 1000.
// ---------------------------------------------------------------------------
__global__ __launch_bounds__(256)
void finalize(const float* __restrict__ bx, const int* __restrict__ keep,
              float* __restrict__ out) {
    for (int r = threadIdx.x; r < POST_N * 4; r += 256) out[PROP_OFF + r] = 0.0f;
    __syncthreads();
    if (threadIdx.x == 0) {
        int cnt = 0;
        for (int i = 0; i < PRE_N && cnt < POST_N; ++i) {
            if (keep[i]) {
                out[PROP_OFF + cnt * 4 + 0] = bx[i];
                out[PROP_OFF + cnt * 4 + 1] = bx[PRE_N + i];
                out[PROP_OFF + cnt * 4 + 2] = bx[2 * PRE_N + i];
                out[PROP_OFF + cnt * 4 + 3] = bx[3 * PRE_N + i];
                ++cnt;
            }
        }
    }
}

// ---------------------------------------------------------------------------
extern "C" void kernel_launch(void* const* d_in, const int* in_sizes, int n_in,
                              void* d_out, int out_size, void* d_ws, size_t ws_size,
                              hipStream_t stream) {
    const float* features = (const float*)d_in[0];
    const float* w_feat   = (const float*)d_in[1];
    const float* b_feat   = (const float*)d_in[2];
    const float* w_obj    = (const float*)d_in[3];
    const float* b_obj    = (const float*)d_in[4];
    const float* w_trans  = (const float*)d_in[5];
    const float* b_trans  = (const float*)d_in[6];
    const int*   im_w     = (const int*)d_in[7];
    const int*   im_h     = (const int*)d_in[8];
    float* out = (float*)d_out;

    char* ws = (char*)d_ws;
    _Float16* F16 = (_Float16*)(ws + 0);            // 77*122*1024*2  = 19,238,912 B
    _Float16* Wp  = (_Float16*)(ws + 19238912);     // 512*9216*2     =  9,437,184 B
    float*    mid = (float*)   (ws + 28676096);     // 9024*512*4     = 18,481,152 B
    float*    key = (float*)   (ws + 47157248);     // 131072*4
    int*      val = (int*)     (ws + 47681536);     // 131072*4
    float*    bx  = (float*)   (ws + 48205824);     // 5*6000*4
    int*      keep= (int*)     (ws + 48325824);     // 6000*4  (end ~48.35 MB)

    // Stage 1: packing + anchors
    {
        int total = HALO_H * HALO_W * C_IN;
        pack_features<<<(total + 255) / 256, 256, 0, stream>>>(features, F16);
    }
    {
        int total = 32 * NQ * 32 * 16;
        pack_weights<<<(total + 255) / 256, 256, 0, stream>>>(w_feat, Wp);
    }
    anchors_kernel<<<(NROWS + 255) / 256, 256, 0, stream>>>(out, im_w, im_h);

    // Stage 2: 3x3 conv via WMMA (141 blocks x 8 waves, 64x64 tile per wave)
    conv3x3_wmma<<<141, 256, 0, stream>>>(F16, Wp, b_feat, mid);

    // Stage 3: 1x1 convs -> objectness / transformers
    conv1x1<<<NPIX, 64, 0, stream>>>(mid, w_obj, b_obj, w_trans, b_trans, out);

    // Stage 4: bitonic sort (descending score, stable tie-break)
    sort_init<<<SORT_N / 256, 256, 0, stream>>>(out, key, val);
    for (int k = 2; k <= SORT_N; k <<= 1)
        for (int j = k >> 1; j > 0; j >>= 1)
            bitonic_step<<<SORT_N / 256, 256, 0, stream>>>(key, val, j, k);

    // Stage 5: transform + clip top 6000
    gather_transform<<<(PRE_N + 255) / 256, 256, 0, stream>>>(val, out, bx, keep,
                                                              im_w, im_h);
    // Stage 6: NMS (sequential semantics)
    nms_kernel<<<1, 1024, 0, stream>>>(bx, keep);

    // Stage 7: compact to 1000 proposals
    finalize<<<1, 256, 0, stream>>>(bx, keep, out);
}
